// MultiHeadAttention_61658550502056
// MI455X (gfx1250) — compile-verified, matrix operand order checked by reading, not run on hardware
//
#include <hip/hip_runtime.h>

// ---------------------------------------------------------------------------
// MHA forward for MI455X (gfx1250, wave32, WMMA bf16).
//   B=4, S=2048, D=1024, H=16, DH=64
// Pipeline:
//   proj_gemm   x3 : q/k/v = split_heads(X@W + b)    -> bf16 [B,H,S,DH] in ws
//   attn_kernel x1 : ctx   = softmax(qk^T/8) v       -> bf16 [B,S,D]    in ws
//   out_gemm    x1 : out   = ctx@Wo + bo             -> fp32 [B,S,D]
// GEMMs: 128x128 WG tile, BK=32, 8 waves x (32x64) => 8 WMMA / wave / strip.
// out_gemm stages its bf16 A-tile with global_load_async_to_lds_b128.
// ---------------------------------------------------------------------------

#define BB   4
#define SS   2048
#define DD   1024
#define HH   16
#define DH   64
#define MM   (BB * SS)   // 8192 rows

typedef __attribute__((ext_vector_type(16))) __bf16 bf16x16;
typedef __attribute__((ext_vector_type(8)))  __bf16 bf16x8;
typedef __attribute__((ext_vector_type(8)))  float  f32x8;

static __device__ __forceinline__ f32x8 wmma_bf16(bf16x16 a, bf16x16 b, f32x8 c) {
  // D = A(16x32 bf16) x B(32x16 bf16) + C(16x16 f32)
  return __builtin_amdgcn_wmma_f32_16x16x32_bf16(false, a, false, b,
                                                 (short)0, c, false, false);
}

// Build a 16-element A/B fragment from two 16-byte chunks (per-lane layout:
// lanes 0-15 hold K {0..7,16..23}, lanes 16-31 hold K {8..15,24..31}).
static __device__ __forceinline__ bf16x16 frag16(const __bf16* p0, const __bf16* p1) {
  bf16x8 lo = *(const bf16x8*)p0;
  bf16x8 hi = *(const bf16x8*)p1;
  bf16x16 r;
#pragma unroll
  for (int i = 0; i < 8; ++i) { r[i] = lo[i]; r[i + 8] = hi[i]; }
  return r;
}

// CDNA5 async copy: 16B global -> LDS, tracked by ASYNCcnt (ISA §15.18.3).
// LDS address = low 32 bits of the generic pointer (aperture truncation).
static __device__ __forceinline__ void async_copy_b128(__bf16* ldsDst,
                                                       const __bf16* gSrc) {
  unsigned l = (unsigned)(uintptr_t)ldsDst;
  asm volatile("global_load_async_to_lds_b128 %0, %1, off"
               :: "v"(l), "v"(gSrc) : "memory");
}

static __device__ __forceinline__ bf16x8 cvt8(float4 a, float4 b) {
  bf16x8 p;
  p[0] = (__bf16)a.x; p[1] = (__bf16)a.y; p[2] = (__bf16)a.z; p[3] = (__bf16)a.w;
  p[4] = (__bf16)b.x; p[5] = (__bf16)b.y; p[6] = (__bf16)b.z; p[7] = (__bf16)b.w;
  return p;
}

// ---------------------------------------------------------------------------
// Projection GEMM: out[b,h,s,dh] = (X @ W + bias) split to heads, bf16 out.
// grid = (DD/128, MM/128) = (8, 64), block = 256
// ---------------------------------------------------------------------------
__global__ __launch_bounds__(256)
void proj_gemm(const float* __restrict__ X, const float* __restrict__ W,
               const float* __restrict__ bias, __bf16* __restrict__ outBHSD) {
  __shared__ __align__(16) __bf16 sA[128][40];  // [m][k]
  __shared__ __align__(16) __bf16 sB[128][40];  // [n][k] (W^T tile)

  const int tid = threadIdx.x;
  const int w = tid >> 5, lane = tid & 31, lr = lane & 15, lh = lane >> 4;
  const int wm = w & 3, wn = w >> 2;            // 4 waves along M, 2 along N
  const int m0 = blockIdx.y * 128, n0 = blockIdx.x * 128;
  const int kb = lh * 8;

  f32x8 acc[2][4] = {};

  for (int k0 = 0; k0 < DD; k0 += 32) {
    __syncthreads();
    {  // A tile 128x32 fp32 -> bf16 LDS; 16 elements / thread
      const int m = tid >> 1, kc = (tid & 1) * 16;
      const float* ap = X + (size_t)(m0 + m) * DD + k0 + kc;
      float4 a0 = *(const float4*)(ap + 0), a1 = *(const float4*)(ap + 4);
      float4 a2 = *(const float4*)(ap + 8), a3 = *(const float4*)(ap + 12);
      *(bf16x8*)&sA[m][kc]     = cvt8(a0, a1);
      *(bf16x8*)&sA[m][kc + 8] = cvt8(a2, a3);
    }
    {  // W tile 32x128 fp32 -> transposed [n][k] bf16; 16 elements / thread
      const int kk = tid >> 3, nb = (tid & 7) * 16;
      const float* bp = W + (size_t)(k0 + kk) * DD + n0 + nb;
      float4 b0 = *(const float4*)(bp + 0), b1 = *(const float4*)(bp + 4);
      float4 b2 = *(const float4*)(bp + 8), b3 = *(const float4*)(bp + 12);
      float t[16] = {b0.x, b0.y, b0.z, b0.w, b1.x, b1.y, b1.z, b1.w,
                     b2.x, b2.y, b2.z, b2.w, b3.x, b3.y, b3.z, b3.w};
#pragma unroll
      for (int j = 0; j < 16; ++j) sB[nb + j][kk] = (__bf16)t[j];
    }
    __syncthreads();

    bf16x16 a0f = frag16(&sA[wm * 32 + lr][kb],      &sA[wm * 32 + lr][kb + 16]);
    bf16x16 a1f = frag16(&sA[wm * 32 + 16 + lr][kb], &sA[wm * 32 + 16 + lr][kb + 16]);
#pragma unroll
    for (int t = 0; t < 4; ++t) {
      const int bn = wn * 64 + t * 16 + lr;
      bf16x16 bf = frag16(&sB[bn][kb], &sB[bn][kb + 16]);
      acc[0][t] = wmma_bf16(a0f, bf, acc[0][t]);
      acc[1][t] = wmma_bf16(a1f, bf, acc[1][t]);
    }
  }

#pragma unroll
  for (int t = 0; t < 4; ++t) {
    const int n_g = n0 + wn * 64 + t * 16 + lr;
    const float bv = bias[n_g];
    const int h = n_g >> 6, dh = n_g & 63;
#pragma unroll
    for (int mt = 0; mt < 2; ++mt) {
#pragma unroll
      for (int j = 0; j < 8; ++j) {
        const int m_g = m0 + wm * 32 + mt * 16 + j + 8 * lh;
        const int b = m_g >> 11, s = m_g & 2047;
        outBHSD[(((size_t)(b * HH + h)) * SS + s) * DH + dh] =
            (__bf16)(acc[mt][t][j] + bv);
      }
    }
  }
}

// ---------------------------------------------------------------------------
// Flash attention: grid = (B*H, S/128), block = 256. Wave w owns 16 queries,
// streams keys in blocks of 64 (8 WMMA for QK^T + 8 WMMA for P*V per block).
// ---------------------------------------------------------------------------
__global__ __launch_bounds__(256)
void attn_kernel(const __bf16* __restrict__ Qh, const __bf16* __restrict__ Kh,
                 const __bf16* __restrict__ Vh, __bf16* __restrict__ Ctx) {
  __shared__ __align__(16) __bf16 sVt[64][72];     // [dh][key]  (V^T tile)
  __shared__ __align__(16) __bf16 sP[8][16][72];   // per-wave P staging

  const int tid = threadIdx.x;
  const int w = tid >> 5, lane = tid & 31, lr = lane & 15, lh = lane >> 4;
  const int bh = blockIdx.x, b = bh >> 4, h = bh & 15;
  const size_t base = (size_t)bh * SS * DH;
  const __bf16* Qp = Qh + base;
  const __bf16* Kp = Kh + base;
  const __bf16* Vp = Vh + base;
  const int q0 = blockIdx.y * 128 + w * 16;

  // Q A-fragments, resident for the whole kernel (2 k-steps over DH=64)
  const __bf16* qrow = Qp + (size_t)(q0 + lr) * DH;
  bf16x16 aq0 = frag16(qrow + lh * 8,      qrow + lh * 8 + 16);
  bf16x16 aq1 = frag16(qrow + 32 + lh * 8, qrow + 32 + lh * 8 + 16);

  f32x8 o[4] = {};
  float mrow[8], lrow[8];
#pragma unroll
  for (int j = 0; j < 8; ++j) { mrow[j] = -1e30f; lrow[j] = 0.0f; }

  for (int kb = 0; kb < SS; kb += 64) {
    __syncthreads();
    {  // stage V block [64 keys][64 dh] -> sVt[dh][key]; 16 elems / thread
      const int key = tid >> 2, db = (tid & 3) * 16;
      bf16x8 v0 = *(const bf16x8*)(Vp + (size_t)(kb + key) * DH + db);
      bf16x8 v1 = *(const bf16x8*)(Vp + (size_t)(kb + key) * DH + db + 8);
#pragma unroll
      for (int j = 0; j < 8; ++j) sVt[db + j][key]     = v0[j];
#pragma unroll
      for (int j = 0; j < 8; ++j) sVt[db + 8 + j][key] = v1[j];
    }
    __syncthreads();

    // scores S[16q x 64keys] = Q x K^T : K rows contiguous over dh ==
    // exactly the B-fragment per-lane layout -> load straight from global.
    f32x8 s[4] = {};
#pragma unroll
    for (int ks = 0; ks < 2; ++ks) {
      const int kc = ks * 32 + lh * 8;
      bf16x16 aq = ks ? aq1 : aq0;
#pragma unroll
      for (int t = 0; t < 4; ++t) {
        const __bf16* kp = Kp + (size_t)(kb + t * 16 + lr) * DH + kc;
        bf16x16 bk = frag16(kp, kp + 16);
        s[t] = wmma_bf16(aq, bk, s[t]);
      }
    }

    // online softmax over this 64-key block; row j lives across 16 lanes
#pragma unroll
    for (int j = 0; j < 8; ++j) {
      float x0 = s[0][j] * 0.125f, x1 = s[1][j] * 0.125f;   // 1/sqrt(64)
      float x2 = s[2][j] * 0.125f, x3 = s[3][j] * 0.125f;
      float mx = fmaxf(fmaxf(x0, x1), fmaxf(x2, x3));
#pragma unroll
      for (int off = 1; off < 16; off <<= 1) mx = fmaxf(mx, __shfl_xor(mx, off, 16));
      const float mnew  = fmaxf(mrow[j], mx);
      const float alpha = __expf(mrow[j] - mnew);
      const float p0 = __expf(x0 - mnew), p1 = __expf(x1 - mnew);
      const float p2 = __expf(x2 - mnew), p3 = __expf(x3 - mnew);
      float rs = (p0 + p1) + (p2 + p3);
#pragma unroll
      for (int off = 1; off < 16; off <<= 1) rs += __shfl_xor(rs, off, 16);
      lrow[j] = lrow[j] * alpha + rs;
      mrow[j] = mnew;
      o[0][j] *= alpha; o[1][j] *= alpha; o[2][j] *= alpha; o[3][j] *= alpha;
      const int r = j + 8 * lh;                 // C-layout row
      sP[w][r][lr]      = (__bf16)p0;
      sP[w][r][16 + lr] = (__bf16)p1;
      sP[w][r][32 + lr] = (__bf16)p2;
      sP[w][r][48 + lr] = (__bf16)p3;
    }
    asm volatile("s_wait_dscnt 0" ::: "memory");  // P stores -> fragment loads

    // O += P[16x64] x V[64x64]
    bf16x16 ap0 = frag16(&sP[w][lr][lh * 8],      &sP[w][lr][lh * 8 + 16]);
    bf16x16 ap1 = frag16(&sP[w][lr][32 + lh * 8], &sP[w][lr][32 + lh * 8 + 16]);
#pragma unroll
    for (int t = 0; t < 4; ++t) {
      bf16x16 bv0 = frag16(&sVt[t * 16 + lr][lh * 8],      &sVt[t * 16 + lr][lh * 8 + 16]);
      bf16x16 bv1 = frag16(&sVt[t * 16 + lr][32 + lh * 8], &sVt[t * 16 + lr][32 + lh * 8 + 16]);
      o[t] = wmma_bf16(ap0, bv0, o[t]);
      o[t] = wmma_bf16(ap1, bv1, o[t]);
    }
  }

  // normalize and write ctx[b, s, h*64 + dh]
  const int hcol = h * DH;
#pragma unroll
  for (int j = 0; j < 8; ++j) {
    const float inv = 1.0f / lrow[j];
    const int q = q0 + j + 8 * lh;
#pragma unroll
    for (int t = 0; t < 4; ++t) {
      Ctx[((size_t)b * SS + q) * DD + hcol + t * 16 + lr] = (__bf16)(o[t][j] * inv);
    }
  }
}

// ---------------------------------------------------------------------------
// Output projection: out = ctx(bf16) @ Wo + bo, fp32 out. Same tiling as proj;
// the bf16 A-tile is staged with CDNA5 async copy (ASYNCcnt) instead of VGPRs.
// ---------------------------------------------------------------------------
__global__ __launch_bounds__(256)
void out_gemm(const __bf16* __restrict__ Xb, const float* __restrict__ W,
              const float* __restrict__ bias, float* __restrict__ out) {
  __shared__ __align__(16) __bf16 sA[128][40];
  __shared__ __align__(16) __bf16 sB[128][40];

  const int tid = threadIdx.x;
  const int w = tid >> 5, lane = tid & 31, lr = lane & 15, lh = lane >> 4;
  const int wm = w & 3, wn = w >> 2;
  const int m0 = blockIdx.y * 128, n0 = blockIdx.x * 128;
  const int kb = lh * 8;

  f32x8 acc[2][4] = {};

  for (int k0 = 0; k0 < DD; k0 += 32) {
    __syncthreads();
    {  // A tile already bf16: async global -> LDS, no VGPR round-trip
      const int m = tid >> 1, kc = (tid & 1) * 16;
      const __bf16* ap = Xb + (size_t)(m0 + m) * DD + k0 + kc;
      async_copy_b128(&sA[m][kc],     ap);
      async_copy_b128(&sA[m][kc + 8], ap + 8);
    }
    {  // W tile 32x128 fp32 -> transposed [n][k] bf16
      const int kk = tid >> 3, nb = (tid & 7) * 16;
      const float* bp = W + (size_t)(k0 + kk) * DD + n0 + nb;
      float4 b0 = *(const float4*)(bp + 0), b1 = *(const float4*)(bp + 4);
      float4 b2 = *(const float4*)(bp + 8), b3 = *(const float4*)(bp + 12);
      float t[16] = {b0.x, b0.y, b0.z, b0.w, b1.x, b1.y, b1.z, b1.w,
                     b2.x, b2.y, b2.z, b2.w, b3.x, b3.y, b3.z, b3.w};
#pragma unroll
      for (int j = 0; j < 16; ++j) sB[nb + j][kk] = (__bf16)t[j];
    }
    asm volatile("s_wait_asynccnt 0x0" ::: "memory");
    __syncthreads();

    bf16x16 a0f = frag16(&sA[wm * 32 + lr][kb],      &sA[wm * 32 + lr][kb + 16]);
    bf16x16 a1f = frag16(&sA[wm * 32 + 16 + lr][kb], &sA[wm * 32 + 16 + lr][kb + 16]);
#pragma unroll
    for (int t = 0; t < 4; ++t) {
      const int bn = wn * 64 + t * 16 + lr;
      bf16x16 bf = frag16(&sB[bn][kb], &sB[bn][kb + 16]);
      acc[0][t] = wmma_bf16(a0f, bf, acc[0][t]);
      acc[1][t] = wmma_bf16(a1f, bf, acc[1][t]);
    }
  }

#pragma unroll
  for (int t = 0; t < 4; ++t) {
    const int n_g = n0 + wn * 64 + t * 16 + lr;
    const float bv = bias[n_g];
#pragma unroll
    for (int mt = 0; mt < 2; ++mt) {
#pragma unroll
      for (int j = 0; j < 8; ++j) {
        const int m_g = m0 + wm * 32 + mt * 16 + j + 8 * lh;
        out[(size_t)m_g * DD + n_g] = acc[mt][t][j] + bv;
      }
    }
  }
}

// ---------------------------------------------------------------------------
extern "C" void kernel_launch(void* const* d_in, const int* in_sizes, int n_in,
                              void* d_out, int out_size, void* d_ws, size_t ws_size,
                              hipStream_t stream) {
  const float* Q  = (const float*)d_in[0];
  const float* K  = (const float*)d_in[1];
  const float* V  = (const float*)d_in[2];
  const float* Wq = (const float*)d_in[3];
  const float* bq = (const float*)d_in[4];
  const float* Wk = (const float*)d_in[5];
  const float* bk = (const float*)d_in[6];
  const float* Wv = (const float*)d_in[7];
  const float* bv = (const float*)d_in[8];
  const float* Wo = (const float*)d_in[9];
  const float* bo = (const float*)d_in[10];
  float* out = (float*)d_out;

  // Workspace: 4 x (B*S*D) bf16 = 64 MiB
  const size_t NE = (size_t)BB * SS * DD;
  __bf16* qh  = (__bf16*)d_ws;
  __bf16* kh  = qh + NE;
  __bf16* vh  = kh + NE;
  __bf16* ctx = vh + NE;

  dim3 gproj(DD / 128, MM / 128);  // (8, 64)
  dim3 gattn(BB * HH, SS / 128);   // (64, 16)

  proj_gemm<<<gproj, 256, 0, stream>>>(Q, Wq, bq, qh);
  proj_gemm<<<gproj, 256, 0, stream>>>(K, Wk, bk, kh);
  proj_gemm<<<gproj, 256, 0, stream>>>(V, Wv, bv, vh);
  attn_kernel<<<gattn, 256, 0, stream>>>(qh, kh, vh, ctx);
  out_gemm<<<gproj, 256, 0, stream>>>(ctx, Wo, bo, out);
}
